// LocalSlidingWindowAttention_25039659336137
// MI455X (gfx1250) — compile-verified, matrix operand order checked by reading, not run on hardware
//
#include <hip/hip_runtime.h>
#include <hip/hip_bf16.h>

typedef __attribute__((ext_vector_type(16))) _Float16 v16h;
typedef __attribute__((ext_vector_type(8)))  _Float16 v8h;
typedef __attribute__((ext_vector_type(4)))  _Float16 v4h;
typedef __attribute__((ext_vector_type(8)))  float    v8f;

#define EN   128      // embed dim
#define NN   8192     // sequence length
#define BB   4        // batch
#define HH   4        // heads
#define DD   32       // head dim
#define WW   33       // window (2*PAD+1)
#define LDH  136      // padded LDS row stride in halves (conflict-free 16B reads)

// Async global->LDS copy of 16 bytes per enabled lane (gfx1250, ASYNCcnt).
__device__ __forceinline__ void async_load_b128(unsigned lds_addr, const void* gptr) {
  asm volatile("global_load_async_to_lds_b128 %0, %1, off"
               :: "v"(lds_addr), "v"(gptr) : "memory");
}
__device__ __forceinline__ void wait_asynccnt0() {
  asm volatile("s_wait_asynccnt 0x0" ::: "memory");
}
// Generic->LDS byte address: low 32 bits of the LDS-aperture address are the
// LDS byte offset (CDNA5 flat-aperture mapping).
__device__ __forceinline__ unsigned lds_addr_of(const void* p) {
  return (unsigned)(size_t)p;
}

// Load a 16x32 f16 WMMA fragment (A layout; B uses identical layout with
// "row" meaning column of the 32x16 B-matrix).
// 16-bit A-matrix layout per CDNA5 ISA 7.12.2:
//   lanes 0-15  hold rows M=0..15, halves 0..7 -> K=0..7,  halves 8..15 -> K=16..23
//   lanes 16-31 hold rows M=0..15, halves 0..7 -> K=8..15, halves 8..15 -> K=24..31
__device__ __forceinline__ v16h load_frag(const _Float16* base, int lane) {
  const int rc = lane & 15;
  const int kb = (lane < 16) ? 0 : 8;
  const _Float16* p = base + rc * LDH + kb;
  v8h lo = *(const v8h*)(p);
  v8h hi = *(const v8h*)(p + 16);
  return __builtin_shufflevector(lo, hi, 0,1,2,3,4,5,6,7,8,9,10,11,12,13,14,15);
}

// ---------------------------------------------------------------------------
// Kernel 1: fused QKV projection.  out = q @ W^T + b for Wq/Wk/Wv.
// Block: 256 threads (8 waves), tile = 64 rows x full 128 cols, K=128.
// Staging keeps the register path (needs f32 -> f16 conversion in flight).
// Writes Q,K,V as f16 in [B,H,N,D] layout.
// ---------------------------------------------------------------------------
__global__ __launch_bounds__(256) void qkv_proj_kernel(
    const float* __restrict__ q,
    const float* __restrict__ Wq, const float* __restrict__ bq,
    const float* __restrict__ Wk, const float* __restrict__ bk,
    const float* __restrict__ Wv, const float* __restrict__ bv,
    _Float16* __restrict__ Qo, _Float16* __restrict__ Ko, _Float16* __restrict__ Vo)
{
  __shared__ alignas(16) _Float16 sA[64 * LDH];
  __shared__ alignas(16) _Float16 sW[128 * LDH];
  const int tid  = threadIdx.x;
  const int lane = tid & 31;
  const int wave = tid >> 5;
  const int row0 = blockIdx.x * 64;

  // Stage 64x128 f32 activations -> f16 LDS
  #pragma unroll
  for (int i = 0; i < 8; ++i) {
    int lin = tid + i * 256;             // float4 index, 2048 total
    int r   = lin >> 5;                  // 32 float4 per row
    int c4  = lin & 31;
    float4 f = *(const float4*)(q + (size_t)(row0 + r) * EN + c4 * 4);
    v4h h = { (_Float16)f.x, (_Float16)f.y, (_Float16)f.z, (_Float16)f.w };
    *(v4h*)(sA + r * LDH + c4 * 4) = h;
  }

  const float* Wmats[3]  = { Wq, Wk, Wv };
  const float* biases[3] = { bq, bk, bv };
  _Float16*    outs[3]   = { Qo, Ko, Vo };

  #pragma unroll
  for (int m = 0; m < 3; ++m) {
    __syncthreads();                      // protect sA (m=0) / sW reuse (m>0)
    const float* W = Wmats[m];
    #pragma unroll
    for (int i = 0; i < 16; ++i) {        // 128x128 f32 weights -> f16 LDS
      int lin = tid + i * 256;
      int r   = lin >> 5;
      int c4  = lin & 31;
      float4 f = *(const float4*)(W + (size_t)r * EN + c4 * 4);
      v4h h = { (_Float16)f.x, (_Float16)f.y, (_Float16)f.z, (_Float16)f.w };
      *(v4h*)(sW + r * LDH + c4 * 4) = h;
    }
    __syncthreads();

    const float* bias = biases[m];
    _Float16* dst = outs[m];
    #pragma unroll
    for (int t = wave; t < 32; t += 8) {  // 4 rowtiles x 8 coltiles
      const int rowtile = t >> 3;
      const int coltile = t & 7;
      v8f c = {0.f,0.f,0.f,0.f,0.f,0.f,0.f,0.f};
      #pragma unroll
      for (int ks = 0; ks < 4; ++ks) {    // K = 128 = 4 x 32
        v16h a = load_frag(sA + rowtile * 16 * LDH + ks * 32, lane);
        v16h b = load_frag(sW + coltile * 16 * LDH + ks * 32, lane);
        c = __builtin_amdgcn_wmma_f32_16x16x32_f16(false, a, false, b,
                                                   (short)0, c, false, false);
      }
      // C/D layout: lane = n + (m>=8 ? 16 : 0), vgpr i = row m%8
      const int col  = coltile * 16 + (lane & 15);
      const float bv_ = bias[col];
      const int mb   = (lane < 16) ? 0 : 8;
      const int hh   = col >> 5;
      const int dd   = col & 31;
      #pragma unroll
      for (int i = 0; i < 8; ++i) {
        const int r  = row0 + rowtile * 16 + mb + i;  // flattened b*N + n
        const int bi = r >> 13;                        // / N
        const int nn = r & (NN - 1);
        dst[(((size_t)(bi * HH + hh)) * NN + nn) * DD + dd] = (_Float16)(c[i] + bv_);
      }
    }
  }
}

// ---------------------------------------------------------------------------
// Kernel 2: sliding-window attention core + softmax.
// One lane per query; K/V window staged in LDS via async global->LDS copies
// (zero halo rows via plain ds stores to disjoint addresses).  Zero-padding
// matches the reference's pad-then-softmax semantics.  Writes probs (f32,
// output 2) and per-head context (f16, [B,N,E]).
// ---------------------------------------------------------------------------
#define KROWS 288   // 256 queries + 2*16 halo
#define SSTR  40    // padded LDS row stride (halves)

__global__ __launch_bounds__(256) void attn_kernel(
    const _Float16* __restrict__ Qf, const _Float16* __restrict__ Kf,
    const _Float16* __restrict__ Vf,
    _Float16* __restrict__ attn, float* __restrict__ probs)
{
  __shared__ alignas(16) _Float16 sK[KROWS * SSTR];
  __shared__ alignas(16) _Float16 sV[KROWS * SSTR];
  const int tid = threadIdx.x;
  const int n0  = blockIdx.x * 256;
  const int bh  = blockIdx.y;            // b*H + h
  const int n   = n0 + tid;
  const size_t headBase = (size_t)bh * NN * DD;
  const _Float16* Kh = Kf + headBase;
  const _Float16* Vh = Vf + headBase;

  // 288 rows x 4 chunks of 16B, for both K and V: one chunk per lane per iter.
  for (int c = tid; c < KROWS * 4; c += 256) {
    const int r  = c >> 2;               // LDS row
    const int q4 = c & 3;                // 16B chunk within row
    const int gn = n0 - 16 + r;          // global key index
    _Float16* kdst = sK + r * SSTR + q4 * 8;
    _Float16* vdst = sV + r * SSTR + q4 * 8;
    if (gn >= 0 && gn < NN) {
      async_load_b128(lds_addr_of(kdst), Kh + (size_t)gn * DD + q4 * 8);
      async_load_b128(lds_addr_of(vdst), Vh + (size_t)gn * DD + q4 * 8);
    } else {
      *(uint4*)kdst = make_uint4(0, 0, 0, 0);
      *(uint4*)vdst = make_uint4(0, 0, 0, 0);
    }
  }
  wait_asynccnt0();
  __syncthreads();

  float qv[32];
  const _Float16* qp = Qf + headBase + (size_t)n * DD;
  #pragma unroll
  for (int i = 0; i < 4; ++i) {
    v8h h = *(const v8h*)(qp + i * 8);
    #pragma unroll
    for (int j = 0; j < 8; ++j) qv[i*8+j] = (float)h[j];
  }

  float sc[WW];
  #pragma unroll
  for (int w = 0; w < WW; ++w) {
    const _Float16* kr = sK + (tid + w) * SSTR;
    float s = 0.f;
    #pragma unroll
    for (int i = 0; i < 4; ++i) {
      v8h h = *(const v8h*)(kr + i * 8);
      #pragma unroll
      for (int j = 0; j < 8; ++j) s = fmaf(qv[i*8+j], (float)h[j], s);
    }
    sc[w] = s * 0.17677669529663687f;    // 1/sqrt(D)
  }

  float mx = sc[0];
  #pragma unroll
  for (int w = 1; w < WW; ++w) mx = fmaxf(mx, sc[w]);
  float sum = 0.f;
  #pragma unroll
  for (int w = 0; w < WW; ++w) { sc[w] = __expf(sc[w] - mx); sum += sc[w]; }
  const float inv = 1.0f / sum;

  float o[32];
  #pragma unroll
  for (int d = 0; d < 32; ++d) o[d] = 0.f;
  #pragma unroll
  for (int w = 0; w < WW; ++w) {
    const float p = sc[w] * inv;
    sc[w] = p;
    const _Float16* vr = sV + (tid + w) * SSTR;
    #pragma unroll
    for (int i = 0; i < 4; ++i) {
      v8h h = *(const v8h*)(vr + i * 8);
      #pragma unroll
      for (int j = 0; j < 8; ++j) o[i*8+j] = fmaf(p, (float)h[j], o[i*8+j]);
    }
  }

  float* pout = probs + ((size_t)bh * NN + n) * WW;
  #pragma unroll
  for (int w = 0; w < WW; ++w) pout[w] = sc[w];

  const int b = bh >> 2;   // / H
  const int h = bh & 3;    // % H
  _Float16* aout = attn + ((size_t)(b * NN + n)) * EN + h * DD;
  #pragma unroll
  for (int i = 0; i < 4; ++i) {
    v8h hv;
    #pragma unroll
    for (int j = 0; j < 8; ++j) hv[j] = (_Float16)o[i*8+j];
    *(v8h*)(aout + i * 8) = hv;
  }
}

// ---------------------------------------------------------------------------
// Kernel 3: output projection.  out = attn @ Wo^T + bo, f32 result.
// A-tile (f16) staged via async global->LDS; Wo staged via register path
// (needs f32 -> f16 conversion).
// ---------------------------------------------------------------------------
__global__ __launch_bounds__(256) void out_proj_kernel(
    const _Float16* __restrict__ attn,
    const float* __restrict__ Wo, const float* __restrict__ bo,
    float* __restrict__ out)
{
  __shared__ alignas(16) _Float16 sA[64 * LDH];
  __shared__ alignas(16) _Float16 sW[128 * LDH];
  const int tid  = threadIdx.x;
  const int lane = tid & 31;
  const int wave = tid >> 5;
  const int row0 = blockIdx.x * 64;

  #pragma unroll
  for (int i = 0; i < 4; ++i) {           // 64x128 f16 -> LDS, async 16B chunks
    int lin = tid + i * 256;              // 1024 chunks
    int r   = lin >> 4;                   // 16 chunks per row
    int c8  = lin & 15;
    _Float16* dst = sA + r * LDH + c8 * 8;
    async_load_b128(lds_addr_of(dst), attn + (size_t)(row0 + r) * EN + c8 * 8);
  }
  #pragma unroll
  for (int i = 0; i < 16; ++i) {          // Wo 128x128 f32 -> f16 LDS
    int lin = tid + i * 256;
    int r   = lin >> 5;
    int c4  = lin & 31;
    float4 f = *(const float4*)(Wo + (size_t)r * EN + c4 * 4);
    v4h h = { (_Float16)f.x, (_Float16)f.y, (_Float16)f.z, (_Float16)f.w };
    *(v4h*)(sW + r * LDH + c4 * 4) = h;
  }
  wait_asynccnt0();
  __syncthreads();

  #pragma unroll
  for (int t = wave; t < 32; t += 8) {
    const int rowtile = t >> 3;
    const int coltile = t & 7;
    v8f c = {0.f,0.f,0.f,0.f,0.f,0.f,0.f,0.f};
    #pragma unroll
    for (int ks = 0; ks < 4; ++ks) {
      v16h a = load_frag(sA + rowtile * 16 * LDH + ks * 32, lane);
      v16h b = load_frag(sW + coltile * 16 * LDH + ks * 32, lane);
      c = __builtin_amdgcn_wmma_f32_16x16x32_f16(false, a, false, b,
                                                 (short)0, c, false, false);
    }
    const int col  = coltile * 16 + (lane & 15);
    const float bval = bo[col];
    const int mb   = (lane < 16) ? 0 : 8;
    #pragma unroll
    for (int i = 0; i < 8; ++i) {
      const int r = row0 + rowtile * 16 + mb + i;
      out[(size_t)r * EN + col] = c[i] + bval;
    }
  }
}

// ---------------------------------------------------------------------------
extern "C" void kernel_launch(void* const* d_in, const int* in_sizes, int n_in,
                              void* d_out, int out_size, void* d_ws, size_t ws_size,
                              hipStream_t stream)
{
  const float* q  = (const float*)d_in[0];
  const float* Wq = (const float*)d_in[1];
  const float* bq = (const float*)d_in[2];
  const float* Wk = (const float*)d_in[3];
  const float* bk = (const float*)d_in[4];
  const float* Wv = (const float*)d_in[5];
  const float* bv = (const float*)d_in[6];
  const float* Wo = (const float*)d_in[7];
  const float* bo = (const float*)d_in[8];

  _Float16* ws = (_Float16*)d_ws;
  const size_t perMat = (size_t)BB * NN * EN;  // 4,194,304 halves = 8 MB each
  _Float16* Qf = ws;
  _Float16* Kf = ws + perMat;
  _Float16* Vf = ws + 2 * perMat;
  _Float16* At = ws + 3 * perMat;              // total 32 MB workspace

  float* out   = (float*)d_out;                // [B,N,E] f32
  float* probs = out + (size_t)BB * NN * EN;   // [B,H,N,W] f32

  qkv_proj_kernel<<<dim3((BB * NN) / 64), 256, 0, stream>>>(
      q, Wq, bq, Wk, bk, Wv, bv, Qf, Kf, Vf);
  attn_kernel<<<dim3(NN / 256, BB * HH), 256, 0, stream>>>(
      Qf, Kf, Vf, At, probs);
  out_proj_kernel<<<dim3((BB * NN) / 64), 256, 0, stream>>>(
      At, Wo, bo, out);
}
